// RNNModel_29961691857232
// MI455X (gfx1250) — matrix operand order repaired and checked by reading, not compiled, not run on hardware
//
#include <hip/hip_runtime.h>

typedef __attribute__((ext_vector_type(16))) _Float16 v16h;
typedef __attribute__((ext_vector_type(8)))  _Float16 v8h;
typedef __attribute__((ext_vector_type(8)))  float    v8f;
typedef __attribute__((ext_vector_type(4)))  float    v4f;

constexpr int Bn = 512, Tn = 512, Dn = 128, Hn = 64, DENSEn = 128, OUTn = 100;
constexpr int K4H  = 4 * Hn;       // 256 gate columns
constexpr int KTOT = Dn + Hn;      // 192 = concat(x, h) K-dim
constexpr int NKB  = KTOT / 32;    // 6 K-blocks of 32 for f16 WMMA
constexpr int XHP  = KTOT + 8;     // padded LDS row stride (200 f16 = 400B, 16B-mult, conflict-free)

__device__ __forceinline__ float sigmoid_f(float x) { return 1.f / (1.f + __expf(-x)); }
__device__ __forceinline__ float tanh_f(float x)    { return 2.f / (1.f + __expf(-2.f * x)) - 1.f; }

__device__ __forceinline__ v8f wmma_f16(v16h a, v16h b, v8f c) {
    // v_wmma_f32_16x16x32_f16: D = A(16x32 f16) * B(32x16 f16) + C(16x16 f32)
    return __builtin_amdgcn_wmma_f32_16x16x32_f16(false, a, false, b, (short)0, c, false, false);
}

// Build the 16-bit A-operand fragment in contiguous K order (same as the B packing):
// lanes 0-15 (M=lane): VGPR0-3 = K 0..7, VGPR4-7 = K 8..15
// lanes 16-31 (M=lane-16): VGPR0-3 = K 16..23, VGPR4-7 = K 24..31
__device__ __forceinline__ v16h load_a_frag(const _Float16* rowp, int kb, int lane) {
    const int half = (lane < 16) ? 0 : 16;
    const v8h lo = *(const v8h*)(rowp + kb * 32 + half);       // K = kb*32 + half .. +7
    const v8h hi = *(const v8h*)(rowp + kb * 32 + 8 + half);   // K = kb*32 + 8 + half .. +7
    v16h a;
#pragma unroll
    for (int e = 0; e < 8; ++e) { a[e] = lo[e]; a[8 + e] = hi[e]; }
    return a;
}

// ---------------------------------------------------------------------------
// Fused LSTM recurrence. One workgroup = 16 batch rows, 4 waves.
// Wave w owns gate-column tiles ct = g*4 + w (g = i,j,f,o) => H-slice [16w,16w+16).
// Cell state c lives in 8 VGPRs per lane for all 512 steps.
// ---------------------------------------------------------------------------
__global__ __launch_bounds__(128) void lstm_fused_kernel(
    const float* __restrict__ seq,   // [B][T][D]
    const int*   __restrict__ slen,  // [B]
    const float* __restrict__ Wl,    // [D+H][4H]
    const float* __restrict__ bl,    // [4H]
    float* __restrict__ last)        // [B][H] out
{
    // B-operand weights, pre-packed per-lane WMMA layout: [ct][kb][lane][16] f16 = 96 KB
    __shared__ __align__(32) _Float16 Wp[16 * NKB * 32 * 16];
    // Double-buffered concat(x_t, h_t): [2][16 rows][XHP] f16
    __shared__ __align__(32) _Float16 xh[2][16][XHP];

    const int tid  = threadIdx.x;
    const int lane = tid & 31;
    const int w    = tid >> 5;            // wave 0..3
    const int row0 = blockIdx.x * 16;
    const int n    = lane & 15;           // output column within tile
    const int msel = (lane >> 4) << 3;    // row offset 0 or 8 (C-tile layout)

    // ---- one-time: pack W_lstm (f32 row-major [K][4H]) into f16 B-frag layout ----
    for (int idx = tid; idx < 16 * NKB * 32; idx += 128) {
        const int ct    = idx / (NKB * 32);
        const int rem   = idx - ct * (NKB * 32);
        const int kb    = rem >> 5;
        const int l     = rem & 31;
        const int col   = ct * 16 + (l & 15);
        const int kbase = kb * 32 + ((l < 16) ? 0 : 16); // B: K contiguous per half-wave
#pragma unroll
        for (int e = 0; e < 16; ++e)
            Wp[idx * 16 + e] = (_Float16)Wl[(kbase + e) * K4H + col];
    }
    for (int i = tid; i < 2 * 16 * XHP; i += 128)
        ((_Float16*)xh)[i] = (_Float16)0.f;  // h_0 = 0 (x region overwritten below)
    __syncthreads();

    // ---- load x_0 into buffer 0 ----
    const int prefr = tid >> 3;            // row 0..15
    const int prefd = (tid & 7) * 16;      // 16 floats per thread
    {
        const float* p = seq + ((size_t)(row0 + prefr) * Tn + 0) * Dn + prefd;
#pragma unroll
        for (int e = 0; e < 16; ++e) xh[0][prefr][prefd + e] = (_Float16)p[e];
    }

    // per-lane gate biases (TF order i, j, f, o; forget bias folded in)
    const float b_i = bl[(0 * 4 + w) * 16 + n];
    const float b_j = bl[(1 * 4 + w) * 16 + n];
    const float b_f = bl[(2 * 4 + w) * 16 + n] + 1.0f; // FORGET_BIAS
    const float b_o = bl[(3 * 4 + w) * 16 + n];

    int sl8[8];
#pragma unroll
    for (int r = 0; r < 8; ++r) sl8[r] = slen[row0 + r + msel];

    v8f c = {};
    __syncthreads();

    for (int t = 0; t < Tn; ++t) {
        const int buf = t & 1, nb = buf ^ 1;

        // prefetch x_{t+1} (non-temporal: streamed once, don't pollute caches)
        v4f xr[4];
        if (t + 1 < Tn) {
            const v4f* p = (const v4f*)(seq + ((size_t)(row0 + prefr) * Tn + (t + 1)) * Dn + prefd);
#pragma unroll
            for (int q = 0; q < 4; ++q) xr[q] = __builtin_nontemporal_load(p + q);
        }

        // z = [x_t, h_t] @ W : K=192 => 6 K-blocks x 4 gates = 24 WMMAs
        v8f zi = {}, zj = {}, zf = {}, zo = {};
        const _Float16* arow = &xh[buf][lane & 15][0];
#pragma unroll
        for (int kb = 0; kb < NKB; ++kb) {
            const v16h a   = load_a_frag(arow, kb, lane);
            const v16h bif = *(const v16h*)&Wp[(((0 * 4 + w) * NKB + kb) * 32 + lane) * 16];
            const v16h bjf = *(const v16h*)&Wp[(((1 * 4 + w) * NKB + kb) * 32 + lane) * 16];
            const v16h bff = *(const v16h*)&Wp[(((2 * 4 + w) * NKB + kb) * 32 + lane) * 16];
            const v16h bof = *(const v16h*)&Wp[(((3 * 4 + w) * NKB + kb) * 32 + lane) * 16];
            zi = wmma_f16(a, bif, zi);
            zj = wmma_f16(a, bjf, zj);
            zf = wmma_f16(a, bff, zf);
            zo = wmma_f16(a, bof, zo);
        }

        // pointwise gates; c stays in registers; h_{t+1} -> LDS next buffer
#pragma unroll
        for (int r = 0; r < 8; ++r) {
            const float ig = sigmoid_f(zi[r] + b_i);
            const float jg = tanh_f(zj[r] + b_j);
            const float fg = sigmoid_f(zf[r] + b_f);
            const float og = sigmoid_f(zo[r] + b_o);
            const float cn = c[r] * fg + ig * jg;
            c[r] = cn;
            const float hv = og * tanh_f(cn);
            const int M = r + msel;
            xh[nb][M][Dn + w * 16 + n] = (_Float16)hv;
            if (t == sl8[r] - 1) last[(row0 + M) * Hn + w * 16 + n] = hv;
        }

        // stage prefetched x_{t+1}
        if (t + 1 < Tn) {
#pragma unroll
            for (int q = 0; q < 4; ++q)
#pragma unroll
                for (int e = 0; e < 4; ++e)
                    xh[nb][prefr][prefd + q * 4 + e] = (_Float16)xr[q][e];
        }
        __syncthreads();
    }
}

// ---------------------------------------------------------------------------
// Head: logits = relu(last @ Wd + bd) @ Wo + bo     (tiny; still WMMA)
// ---------------------------------------------------------------------------
constexpr int LP = Hn + 8;        // 72 f16 = 144B row stride (16B multiple)
constexpr int DP = DENSEn + 8;    // 136 f16 = 272B
constexpr int OTILES = (OUTn + 15) / 16; // 7

__global__ __launch_bounds__(128) void head_kernel(
    const float* __restrict__ last,  // [B][H]
    const float* __restrict__ Wd,    // [H][DENSE]
    const float* __restrict__ bd,    // [DENSE]
    const float* __restrict__ Wo,    // [DENSE][OUT]
    const float* __restrict__ bo,    // [OUT]
    float* __restrict__ out)         // [B][OUT]
{
    __shared__ __align__(32) _Float16 Wdp[8 * 2 * 32 * 16];       // 16 KB
    __shared__ __align__(32) _Float16 Wop[OTILES * 4 * 32 * 16];  // 28 KB
    __shared__ __align__(32) _Float16 lastA[16][LP];
    __shared__ __align__(32) _Float16 denseA[16][DP];

    const int tid = threadIdx.x, lane = tid & 31, w = tid >> 5;
    const int row0 = blockIdx.x * 16;
    const int n = lane & 15;
    const int msel = (lane >> 4) << 3;

    // pack W_dense
    for (int idx = tid; idx < 8 * 2 * 32; idx += 128) {
        const int ct = idx / 64, rem = idx - ct * 64;
        const int kb = rem >> 5, l = rem & 31;
        const int col = ct * 16 + (l & 15);
        const int kbase = kb * 32 + ((l < 16) ? 0 : 16);
#pragma unroll
        for (int e = 0; e < 16; ++e)
            Wdp[idx * 16 + e] = (_Float16)Wd[(kbase + e) * DENSEn + col];
    }
    // pack W_out (zero-pad cols >= 100)
    for (int idx = tid; idx < OTILES * 4 * 32; idx += 128) {
        const int ct = idx / 128, rem = idx - ct * 128;
        const int kb = rem >> 5, l = rem & 31;
        const int col = ct * 16 + (l & 15);
        const int kbase = kb * 32 + ((l < 16) ? 0 : 16);
#pragma unroll
        for (int e = 0; e < 16; ++e)
            Wop[idx * 16 + e] = (col < OUTn) ? (_Float16)Wo[(kbase + e) * OUTn + col]
                                             : (_Float16)0.f;
    }
    for (int i = tid; i < 16 * Hn; i += 128)
        lastA[i >> 6][i & 63] = (_Float16)last[(row0 + (i >> 6)) * Hn + (i & 63)];
    __syncthreads();

    // dense layer: wave w handles column tiles {w, w+4}; K = 64 => 2 WMMAs each
#pragma unroll
    for (int s = 0; s < 2; ++s) {
        const int ct = w + s * 4;
        v8f acc = {};
        const _Float16* arow = &lastA[lane & 15][0];
#pragma unroll
        for (int kb = 0; kb < 2; ++kb) {
            const v16h a  = load_a_frag(arow, kb, lane);
            const v16h bf = *(const v16h*)&Wdp[((ct * 2 + kb) * 32 + lane) * 16];
            acc = wmma_f16(a, bf, acc);
        }
        const int col = ct * 16 + n;
        const float bb = bd[col];
#pragma unroll
        for (int r = 0; r < 8; ++r) {
            float v = acc[r] + bb;
            v = v > 0.f ? v : 0.f;
            denseA[r + msel][col] = (_Float16)v;
        }
    }
    __syncthreads();

    // output layer: 7 tiles over 4 waves; K = 128 => 4 WMMAs each
    for (int ct = w; ct < OTILES; ct += 4) {
        v8f acc = {};
        const _Float16* arow = &denseA[lane & 15][0];
#pragma unroll
        for (int kb = 0; kb < 4; ++kb) {
            const v16h a  = load_a_frag(arow, kb, lane);
            const v16h bf = *(const v16h*)&Wop[((ct * 4 + kb) * 32 + lane) * 16];
            acc = wmma_f16(a, bf, acc);
        }
        const int col = ct * 16 + n;
        if (col < OUTn) {
            const float bb = bo[col];
#pragma unroll
            for (int r = 0; r < 8; ++r)
                out[(row0 + r + msel) * OUTn + col] = acc[r] + bb;
        }
    }
}

extern "C" void kernel_launch(void* const* d_in, const int* in_sizes, int n_in,
                              void* d_out, int out_size, void* d_ws, size_t ws_size,
                              hipStream_t stream) {
    (void)in_sizes; (void)n_in; (void)out_size; (void)ws_size;
    const float* seq  = (const float*)d_in[0];
    const int*   slen = (const int*)  d_in[1];
    const float* Wl   = (const float*)d_in[2];
    const float* bl   = (const float*)d_in[3];
    const float* Wd   = (const float*)d_in[4];
    const float* bd   = (const float*)d_in[5];
    const float* Wo   = (const float*)d_in[6];
    const float* bo   = (const float*)d_in[7];
    float* out  = (float*)d_out;
    float* lastbuf = (float*)d_ws;  // [512][64] f32 = 128 KB scratch

    lstm_fused_kernel<<<Bn / 16, 128, 0, stream>>>(seq, slen, Wl, bl, lastbuf);
    head_kernel<<<Bn / 16, 128, 0, stream>>>(lastbuf, Wd, bd, Wo, bo, out);
}
